// LongAttentionHead_15839839388309
// MI455X (gfx1250) — compile-verified, matrix operand order checked by reading, not run on hardware
//
#include <hip/hip_runtime.h>
#include <hip/hip_bf16.h>

// Problem constants (from reference): B=2, S=4096, E=1024, D=64, W=129.
#define BB 2
#define SS 4096
#define EE 1024
#define DD 64
#define HALF_W 64            // window half-width: W=129 -> [q-64, q+64]
#define NKEY 160             // padded key window per 16-query block
#define KOFF0 72             // key window base = s0 - 72
#define VTS (SS + NKEY)      // padded seq length for K/V halo layouts (4256)
#define INV_SQRT_E 0.03125f  // 1/sqrt(1024)

typedef __attribute__((ext_vector_type(16))) __bf16 bf16x16;
typedef __attribute__((ext_vector_type(8)))  float  f32x8;
typedef __attribute__((ext_vector_type(4)))  unsigned int u32x4;
typedef __attribute__((ext_vector_type(8)))  int i32x8;
typedef __attribute__((ext_vector_type(4)))  int i32x4;

static __device__ __forceinline__ f32x8 wmma_bf16(bf16x16 a, bf16x16 b, f32x8 c) {
    return __builtin_amdgcn_wmma_f32_16x16x32_bf16(false, a, false, b, (short)0, c,
                                                   false, false);
}

static __device__ __forceinline__ void ld8(const __bf16* p, bf16x16& d, int o) {
    uint4 u = *reinterpret_cast<const uint4*>(p);
    const __bf16* h = reinterpret_cast<const __bf16*>(&u);
#pragma unroll
    for (int j = 0; j < 8; ++j) d[o + j] = h[j];
}

// 16-bit 16x32 A / 32x16 B fragment (ISA 7.12.2): lane l<16 holds major index l,
// K = k0 + {0..7, 16..23}; lane l+16: K = k0 + {8..15, 24..31}.
static __device__ __forceinline__ bf16x16 frag_rows(const __bf16* laneRow, int k0) {
    int lane = threadIdx.x & 31;
    int koff = k0 + ((lane >> 4) << 3);
    bf16x16 f;
    ld8(laneRow + koff, f, 0);
    ld8(laneRow + koff + 16, f, 8);
    return f;
}

static __device__ __forceinline__ void cvt8(bf16x16& d, int o, float4 a, float4 b) {
    d[o + 0] = (__bf16)a.x; d[o + 1] = (__bf16)a.y;
    d[o + 2] = (__bf16)a.z; d[o + 3] = (__bf16)a.w;
    d[o + 4] = (__bf16)b.x; d[o + 5] = (__bf16)b.y;
    d[o + 6] = (__bf16)b.z; d[o + 7] = (__bf16)b.w;
}

// ---------------------------------------------------------------------------
// Tensor Data Mover: 2D tile load Global->LDS (D# per ISA cdna5 8.3/8.4).
// data_size = 2 bytes (bf16). tile_d1 lines of tile_d0 elements, line stride
// stride0 (elements). Groups 2/3 zero => dims 2..4 unused.
// ---------------------------------------------------------------------------
#if __has_builtin(__builtin_amdgcn_tensor_load_to_lds)
#define HAVE_TDM 1
static __device__ __forceinline__ void tdm_load_2d(unsigned lds_byte_off,
                                                   const void* gaddr,
                                                   unsigned tile_d0, unsigned tile_d1,
                                                   unsigned long long stride0,
                                                   unsigned tensor_d0,
                                                   unsigned tensor_d1) {
    unsigned long long ga = (unsigned long long)gaddr;
    u32x4 g0;
    g0[0] = 1u;                                    // count=1, user mode, no gather
    g0[1] = lds_byte_off;                          // lds_addr [63:32]
    g0[2] = (unsigned)(ga & 0xffffffffu);          // global_addr[31:0]
    g0[3] = (unsigned)((ga >> 32) & 0x1ffffffu)    // global_addr[56:32]
          | (2u << 30);                            // type=2 ("image")
    i32x8 g1;
    g1[0] = (int)(1u << 16);                       // data_size=1 (2B); mask=0
    g1[1] = (int)((tensor_d0 & 0xffffu) << 16);    // tensor_dim0[15:0] @ bits 63:48
    g1[2] = (int)(((tensor_d0 >> 16) & 0xffffu) | ((tensor_d1 & 0xffffu) << 16));
    g1[3] = (int)(((tensor_d1 >> 16) & 0xffffu) | ((tile_d0 & 0xffffu) << 16));
    g1[4] = (int)(tile_d1 & 0xffffu);              // tile_dim1; tile_dim2=0
    g1[5] = (int)(unsigned)(stride0 & 0xffffffffu);        // dim0 stride lo
    g1[6] = (int)(unsigned)((stride0 >> 32) & 0xffffu);    // dim0 stride hi; stride1=0
    g1[7] = 0;
    i32x4 z4 = {0, 0, 0, 0};
#if __clang_major__ >= 23
    i32x8 z8 = {0, 0, 0, 0, 0, 0, 0, 0};
    __builtin_amdgcn_tensor_load_to_lds(g0, g1, z4, z4, z8, 0);
#else
    __builtin_amdgcn_tensor_load_to_lds(g0, g1, z4, z4, 0);
#endif
}
static __device__ __forceinline__ unsigned lds_off(const void* p) {
    return (unsigned)(unsigned long long)(size_t)p;   // low 32 bits = LDS offset
}
#else
#define HAVE_TDM 0
#endif

// ---------------------------------------------------------------------------
// Kernel 1a: fp32 -> bf16 weight conversion WITH transpose: Wt[3][D][E].
// ---------------------------------------------------------------------------
__global__ void __launch_bounds__(256) cvt_weights_t(const float* __restrict__ wq,
                                                     const float* __restrict__ wk,
                                                     const float* __restrict__ wv,
                                                     unsigned short* __restrict__ wt_) {
    __bf16* wt = reinterpret_cast<__bf16*>(wt_);
    int i = blockIdx.x * 256 + threadIdx.x;
    const int n = DD * EE;
    if (i >= 3 * n) return;
    const int m = i / n, r = i % n;
    const int d = r / EE, e = r % EE;
    const float* src = (m == 0) ? wq : (m == 1) ? wk : wv;
    wt[i] = (__bf16)src[e * DD + d];
}

// ---------------------------------------------------------------------------
// Kernel 1b: zero the halos of Kpad[B][VTS][D] and Vt[B][D][VTS]
// (rows [0,72) and [72+S, VTS)). Haloed tiles keep all TDM loads in-bounds.
// ---------------------------------------------------------------------------
__global__ void __launch_bounds__(256) zero_halos(unsigned short* __restrict__ kp_,
                                                  unsigned short* __restrict__ vt_) {
    __bf16* kp = reinterpret_cast<__bf16*>(kp_);
    __bf16* vt = reinterpret_cast<__bf16*>(vt_);
    const int per = VTS - SS;                     // 160 halo rows/cols per line
    const int nk = BB * per * DD;                 // Kpad halo elements
    const int nv = BB * DD * per;                 // Vt halo elements
    int i = blockIdx.x * 256 + threadIdx.x;
    if (i < nk) {
        const int bp = i / (per * DD);
        const int rr = (i / DD) % per;
        const int cc = i % DD;
        const int row = (rr < KOFF0) ? rr : (rr + SS);
        kp[((size_t)bp * VTS + row) * DD + cc] = (__bf16)0.f;
    } else if (i < nk + nv) {
        const int j = i - nk;
        const int bd = j / per, p = j % per;
        const int idx = (p < KOFF0) ? p : (p + SS);
        vt[(size_t)bd * VTS + idx] = (__bf16)0.f;
    }
}

// ---------------------------------------------------------------------------
// Kernel 2: fused QKV projection (WMMA bf16, f32 accum). Q row-major;
// K into padded Kpad[B][VTS][D]; V transposed+padded Vt[B][D][VTS].
// ---------------------------------------------------------------------------
__global__ void __launch_bounds__(128) qkv_proj(const float* __restrict__ x,
                                                const unsigned short* __restrict__ wt_,
                                                const float* __restrict__ bq,
                                                const float* __restrict__ bk,
                                                const float* __restrict__ bv,
                                                unsigned short* __restrict__ qb_,
                                                unsigned short* __restrict__ kp_,
                                                unsigned short* __restrict__ vt_) {
    const __bf16* wt = reinterpret_cast<const __bf16*>(wt_);
    __bf16* qb = reinterpret_cast<__bf16*>(qb_);
    __bf16* kp = reinterpret_cast<__bf16*>(kp_);
    __bf16* vt = reinterpret_cast<__bf16*>(vt_);

    const int row0 = blockIdx.x * 16;
    const int wave = threadIdx.x >> 5;
    const int lane = threadIdx.x & 31;
    const int l    = lane & 15;
    const int half = lane >> 4;
    const int col  = wave * 16 + l;

    const float*  xrow = x + (size_t)(row0 + l) * EE;
    const __bf16* wqc  = wt + (size_t)(0 * DD + col) * EE;
    const __bf16* wkc  = wt + (size_t)(1 * DD + col) * EE;
    const __bf16* wvc  = wt + (size_t)(2 * DD + col) * EE;

    f32x8 aq = {}, ak = {}, av = {};

    for (int k0 = 0; k0 < EE; k0 += 32) {
        const int koff = k0 + (half << 3);
        bf16x16 a;
        cvt8(a, 0, *(const float4*)(xrow + koff),      *(const float4*)(xrow + koff + 4));
        cvt8(a, 8, *(const float4*)(xrow + koff + 16), *(const float4*)(xrow + koff + 20));
        bf16x16 fq = frag_rows(wqc, k0);
        bf16x16 fk = frag_rows(wkc, k0);
        bf16x16 fv = frag_rows(wvc, k0);
        aq = wmma_bf16(a, fq, aq);
        ak = wmma_bf16(a, fk, ak);
        av = wmma_bf16(a, fv, av);
        if (k0 + 32 < EE) __builtin_prefetch(xrow + koff + 32, 0, 3);
    }

    const float biq = bq[col], bik = bk[col], biv = bv[col];
    const int b  = row0 / SS;
    const int sl = row0 % SS;

#pragma unroll
    for (int r = 0; r < 8; ++r) {
        const int M = r + (half << 3);
        qb[(size_t)(row0 + M) * DD + col] = (__bf16)(aq[r] + biq);
        kp[((size_t)b * VTS + KOFF0 + sl + M) * DD + col] = (__bf16)(ak[r] + bik);
    }
    union { uint4 u; __bf16 h[8]; } t;
#pragma unroll
    for (int r = 0; r < 8; ++r) t.h[r] = (__bf16)(av[r] + biv);
    *reinterpret_cast<uint4*>(vt + (size_t)(b * DD + col) * VTS + KOFF0 + sl +
                              (half << 3)) = t.u;
}

// ---------------------------------------------------------------------------
// Kernel 3: sliding-window attention. K tile (160x64, contiguous 20KB) and
// V tile (64 lines x 160, stride VTS) staged into LDS by the Tensor Data
// Mover (s_wait_tensorcnt + barrier), then K/V/P WMMA fragments are LDS b128
// reads. Scores -> masked softmax -> P.V, all 16x16x32 bf16 WMMAs.
// ---------------------------------------------------------------------------
__global__ void __launch_bounds__(128) sliding_attn(const unsigned short* __restrict__ qb_,
                                                    const unsigned short* __restrict__ kp_,
                                                    const unsigned short* __restrict__ vt_,
                                                    float* __restrict__ out) {
    const __bf16* qb = reinterpret_cast<const __bf16*>(qb_);
    const __bf16* kp = reinterpret_cast<const __bf16*>(kp_);
    const __bf16* vt = reinterpret_cast<const __bf16*>(vt_);

    __shared__ __attribute__((aligned(16))) __bf16 sK[NKEY * DD];  // [win][d]
    __shared__ __attribute__((aligned(16))) __bf16 sV[DD * NKEY];  // [d][win]
    __shared__ __attribute__((aligned(16))) float  sS[16 * NKEY];
    __shared__ __attribute__((aligned(16))) __bf16 sP[16 * NKEY];
    __shared__ float sRed[16][8];

    const int blk  = blockIdx.x;
    const int b    = blk / (SS / 16);
    const int s0   = (blk % (SS / 16)) * 16;
    const int wave = threadIdx.x >> 5;
    const int lane = threadIdx.x & 31;
    const int l    = lane & 15;
    const int half = lane >> 4;
    const int kbase = s0 - KOFF0;

    // ---- Stage K/V window tiles into LDS (TDM; fallback: cooperative copy)
#if HAVE_TDM
    if (threadIdx.x < 32) {
        // K: one contiguous line of NKEY*DD bf16 starting at padded row s0
        tdm_load_2d(lds_off(sK), kp + ((size_t)b * VTS + s0) * DD,
                    NKEY * DD, 1, NKEY * DD, NKEY * DD, 1);
        // V: DD lines of NKEY bf16, line stride VTS
        tdm_load_2d(lds_off(sV), vt + (size_t)(b * DD) * VTS + s0,
                    NKEY, DD, VTS, VTS, DD);
        __builtin_amdgcn_s_wait_tensorcnt(0);
    }
#else
    for (int i = threadIdx.x; i < NKEY * DD / 8; i += 128)
        ((uint4*)sK)[i] = *(const uint4*)(kp + ((size_t)b * VTS + s0) * DD + i * 8);
    for (int i = threadIdx.x; i < DD * NKEY / 8; i += 128) {
        const int line = i / (NKEY / 8), seg = i % (NKEY / 8);
        ((uint4*)sV)[i] =
            *(const uint4*)(vt + (size_t)(b * DD + line) * VTS + s0 + seg * 8);
    }
#endif
    const __bf16* qrow = qb + (size_t)(b * SS + s0 + l) * DD;
    __syncthreads();

    // ---- Phase 1: scores = Q . K^T from LDS (wave-uniform loop, EXEC full)
    for (int t = wave; t < NKEY / 16; t += 4) {
        const __bf16* krow = sK + (t * 16 + l) * DD;
        f32x8 acc = {};
#pragma unroll
        for (int k0 = 0; k0 < DD; k0 += 32) {
            bf16x16 a   = frag_rows(qrow, k0);
            bf16x16 bfr = frag_rows(krow, k0);
            acc = wmma_bf16(a, bfr, acc);
        }
#pragma unroll
        for (int r = 0; r < 8; ++r) {
            const int M    = r + (half << 3);
            const int n    = t * 16 + l;
            const int q    = s0 + M;
            const int kpos = kbase + n;
            const bool valid = (kpos >= q - HALF_W) && (kpos <= q + HALF_W) &&
                               (kpos >= 0) && (kpos < SS);
            sS[M * NKEY + n] = valid ? acc[r] * INV_SQRT_E : -__builtin_inff();
        }
    }
    __syncthreads();

    // ---- Phase 2: softmax, 8 threads per row x 20 cols each
    {
        const int row = threadIdx.x >> 3;
        const int sub = threadIdx.x & 7;
        const int c0  = sub * (NKEY / 8);
        float m = -__builtin_inff();
        for (int c = c0; c < c0 + NKEY / 8; ++c) m = fmaxf(m, sS[row * NKEY + c]);
        sRed[row][sub] = m;
        __syncthreads();
        float rm = sRed[row][0];
#pragma unroll
        for (int j = 1; j < 8; ++j) rm = fmaxf(rm, sRed[row][j]);
        __syncthreads();
        float s = 0.f;
        for (int c = c0; c < c0 + NKEY / 8; ++c) {
            float e = __expf(sS[row * NKEY + c] - rm);
            sS[row * NKEY + c] = e;
            s += e;
        }
        sRed[row][sub] = s;
        __syncthreads();
        float rs = 0.f;
#pragma unroll
        for (int j = 0; j < 8; ++j) rs += sRed[row][j];
        const float inv = 1.f / rs;
        for (int c = c0; c < c0 + NKEY / 8; ++c)
            sP[row * NKEY + c] = (__bf16)(sS[row * NKEY + c] * inv);
    }
    __syncthreads();

    // ---- Phase 3: out = P . V from LDS (wave i -> output cols [16i,16i+16))
    const int d = wave * 16 + l;
    const __bf16* prow = sP + l * NKEY;
    const __bf16* vline = sV + d * NKEY;
    f32x8 acc = {};
#pragma unroll
    for (int k0 = 0; k0 < NKEY; k0 += 32) {
        bf16x16 a   = frag_rows(prow, k0);
        bf16x16 bfr = frag_rows(vline, k0);
        acc = wmma_bf16(a, bfr, acc);
    }
#pragma unroll
    for (int r = 0; r < 8; ++r) {
        const int M = r + (half << 3);
        out[(size_t)(b * SS + s0 + M) * DD + d] = acc[r];
    }
}

// ---------------------------------------------------------------------------
// Launch: inputs = {x, Wq, bq, Wk, bk, Wv, bv, window_size}
// Workspace (bf16): [Wt 3*D*E][Q B*S*D][Kpad B*VTS*D][Vt B*D*VTS] (~3.5 MB)
// ---------------------------------------------------------------------------
extern "C" void kernel_launch(void* const* d_in, const int* in_sizes, int n_in,
                              void* d_out, int out_size, void* d_ws, size_t ws_size,
                              hipStream_t stream) {
    const float* x  = (const float*)d_in[0];
    const float* Wq = (const float*)d_in[1];
    const float* bq = (const float*)d_in[2];
    const float* Wk = (const float*)d_in[3];
    const float* bk = (const float*)d_in[4];
    const float* Wv = (const float*)d_in[5];
    const float* bv = (const float*)d_in[6];

    unsigned short* wsu = (unsigned short*)d_ws;
    unsigned short* wt  = wsu;                        // 3*D*E
    unsigned short* qbp = wt + 3 * DD * EE;           // B*S*D
    unsigned short* kpp = qbp + BB * SS * DD;         // B*VTS*D
    unsigned short* vtp = kpp + BB * VTS * DD;        // B*D*VTS

    cvt_weights_t<<<(3 * DD * EE + 255) / 256, 256, 0, stream>>>(Wq, Wk, Wv, wt);
    zero_halos<<<(2 * BB * (VTS - SS) * DD + 255) / 256, 256, 0, stream>>>(kpp, vtp);
    qkv_proj<<<(BB * SS) / 16, 128, 0, stream>>>(x, wt, bq, bk, bv, qbp, kpp, vtp);
    sliding_attn<<<(BB * SS) / 16, 128, 0, stream>>>(qbp, kpp, vtp, (float*)d_out);
}
